// AutoregressiveTransformer_80341658239311
// MI455X (gfx1250) — matrix operand order repaired and checked by reading, not compiled, not run on hardware
//
#include <hip/hip_runtime.h>
#include <hip/hip_bf16.h>

typedef __attribute__((ext_vector_type(16))) _Float16 v16h;
typedef __attribute__((ext_vector_type(8)))  _Float16 v8h;
typedef __attribute__((ext_vector_type(4)))  _Float16 v4h;
typedef __attribute__((ext_vector_type(8)))  float    v8f;
typedef __attribute__((ext_vector_type(4)))  float    v4f;

// Model constants (from reference)
#define BB   16
#define TT   1024
#define CC   128
#define HH   4
#define HS   32
#define FF   512
#define LL   2
#define VV   10000
#define BT   (BB*TT)

static __device__ __forceinline__ v8f wmma_f16(v16h a, v16h b, v8f c) {
    // D = A(16x32 f16) * B(32x16 f16) + C(16x16 f32)
    return __builtin_amdgcn_wmma_f32_16x16x32_f16(false, a, false, b, (short)0, c, false, false);
}

union FragH { v16h v; v8h h[2]; };

// ---------------------------------------------------------------------------
// Weight convert + transpose:  dst[N,K] f16  <-  src[K,N] f32
// ---------------------------------------------------------------------------
__global__ void convt_kernel(const float* __restrict__ src, _Float16* __restrict__ dst,
                             int K, int N) {
    int i = blockIdx.x * blockDim.x + threadIdx.x;
    if (i >= K * N) return;
    int n = i / K, k = i % K;
    dst[i] = (_Float16)src[k * N + n];
}

// ---------------------------------------------------------------------------
// Embedding: x[bt,c] = tok_emb[idx[bt],c] + pos_emb[t,c]   (f32)
// ---------------------------------------------------------------------------
__global__ void embed_kernel(const int* __restrict__ idx,
                             const float* __restrict__ tok_emb,
                             const float* __restrict__ pos_emb,
                             float* __restrict__ x) {
    int i = blockIdx.x * blockDim.x + threadIdx.x;          // BT*32 threads
    if (i >= BT * 32) return;
    int bt = i >> 5;
    int c4 = (i & 31) << 2;
    int t  = bt & (TT - 1);
    int tok = idx[bt];
    v4f te = *(const v4f*)(tok_emb + (size_t)tok * CC + c4);
    v4f pe = *(const v4f*)(pos_emb + (size_t)t   * CC + c4);
    *(v4f*)(x + (size_t)bt * CC + c4) = te + pe;
}

// ---------------------------------------------------------------------------
// LayerNorm over C=128, one wave per row, output f16
// ---------------------------------------------------------------------------
__global__ __launch_bounds__(256) void layernorm_kernel(
        const float* __restrict__ x, const float* __restrict__ g,
        const float* __restrict__ b, _Float16* __restrict__ out, int rows) {
    int row  = blockIdx.x * (blockDim.x >> 5) + (threadIdx.x >> 5);
    if (row >= rows) return;
    int lane = threadIdx.x & 31;
    v4f xv = *(const v4f*)(x + (size_t)row * CC + lane * 4);
    float s  = xv[0] + xv[1] + xv[2] + xv[3];
    float ss = xv[0]*xv[0] + xv[1]*xv[1] + xv[2]*xv[2] + xv[3]*xv[3];
    for (int m = 16; m >= 1; m >>= 1) { s += __shfl_xor(s, m, 32); ss += __shfl_xor(ss, m, 32); }
    float mean = s * (1.f / CC);
    float var  = ss * (1.f / CC) - mean * mean;
    float rstd = rsqrtf(var + 1e-5f);
    v4h o;
    for (int j = 0; j < 4; j++) {
        int c = lane * 4 + j;
        o[j] = (_Float16)((xv[j] - mean) * rstd * g[c] + b[c]);
    }
    *(v4h*)(out + (size_t)row * CC + lane * 4) = o;
}

// ---------------------------------------------------------------------------
// Generic WMMA GEMM:  out[M,N] = A[M,K](f16) * Bt[N,K]^T(f16) (+bias)(+resid)
// One wave computes a 16x64 tile (4 accumulators). M%16==0, N%16==0, K%32==0.
// OMODE: 0=f32 out, 1=f16 out, 2=f16 transposed out (vt[(m/Tld)*N+n][Tld]+t)
// NT: non-temporal stores (streaming output larger than L2, never re-read)
// ---------------------------------------------------------------------------
enum { OUT_F32 = 0, OUT_F16 = 1, OUT_F16T = 2 };

template<int OMODE, bool RELU, bool RESID, bool NT>
__global__ __launch_bounds__(256) void gemm16_kernel(
        const _Float16* __restrict__ A, const _Float16* __restrict__ Bt,
        const float* __restrict__ bias, const float* __restrict__ resid,
        void* __restrict__ outp, int M, int N, int K, int Tld) {
    const int lane = threadIdx.x & 31;
    const int half = lane >> 4;
    const int sub  = lane & 15;
    int wid = blockIdx.x * (blockDim.x >> 5) + (threadIdx.x >> 5);
    int Nt = N >> 4;
    int Ng = (Nt + 3) >> 2;
    if (wid >= (M >> 4) * Ng) return;                 // wave-uniform
    int mt = wid / Ng, ng = wid % Ng;
    int mb = mt << 4;

    v8f acc[4];
    for (int i = 0; i < 4; i++) for (int j = 0; j < 8; j++) acc[i][j] = 0.f;

    const _Float16* arow = A + (size_t)(mb + sub) * K;
    for (int kb = 0; kb < K; kb += 32) {
        FragH a;                                      // A 16x32 layout (ISA 7.12.2)
        a.h[0] = *(const v8h*)(arow + kb + half * 8);
        a.h[1] = *(const v8h*)(arow + kb + 16 + half * 8);
        for (int nt = 0; nt < 4; nt++) {
            int n0 = (ng * 4 + nt) << 4;
            if (n0 >= N) break;                       // wave-uniform
            FragH bf;                                 // B 32x16: lane=col, K contiguous
            const _Float16* brow = Bt + (size_t)(n0 + sub) * K + kb + half * 8;
            bf.h[0] = *(const v8h*)(brow);
            bf.h[1] = *(const v8h*)(brow + 16);
            acc[nt] = wmma_f16(a.v, bf.v, acc[nt]);
        }
    }

    for (int nt = 0; nt < 4; nt++) {
        int n0 = (ng * 4 + nt) << 4;
        if (n0 >= N) break;
        int col = n0 + sub;
        float bs = bias ? bias[col] : 0.f;
        if constexpr (OMODE == OUT_F16T) {
            v8h pk;
            for (int r = 0; r < 8; r++) {
                float v = acc[nt][r] + bs;
                if (RELU) v = fmaxf(v, 0.f);
                pk[r] = (_Float16)v;
            }
            int m0 = mb + half * 8;                   // 8 consecutive rows -> contiguous t
            _Float16* vt = (_Float16*)outp;
            *(v8h*)(vt + ((size_t)(m0 / Tld) * N + col) * Tld + (m0 % Tld)) = pk;
        } else {
            for (int r = 0; r < 8; r++) {
                size_t idx = (size_t)(mb + half * 8 + r) * N + col;
                float v = acc[nt][r] + bs;
                if (RELU) v = fmaxf(v, 0.f);
                if constexpr (RESID) v += resid[idx];
                if constexpr (OMODE == OUT_F32) {
                    float* o = (float*)outp;
                    if constexpr (NT) __builtin_nontemporal_store(v, o + idx);
                    else              o[idx] = v;
                } else {
                    ((_Float16*)outp)[idx] = (_Float16)v;
                }
            }
        }
    }
}

// ---------------------------------------------------------------------------
// Causal flash attention. One wave per (b, h, 16-query tile). hs=32 = WMMA K.
// q,k: [BT, C] f16 (head h at cols h*32..). vt: [B*C, T] f16 (V transposed).
// att: [BT, C] f16 output.
// ---------------------------------------------------------------------------
__global__ __launch_bounds__(256) void attn_kernel(
        const _Float16* __restrict__ q, const _Float16* __restrict__ kk,
        const _Float16* __restrict__ vt, _Float16* __restrict__ att) {
    __shared__ _Float16 lds[8][16 * 32];              // per-wave P transpose tile
    const int lane = threadIdx.x & 31;
    const int half = lane >> 4;
    const int sub  = lane & 15;
    const int w    = threadIdx.x >> 5;
    int gid = blockIdx.x * 8 + w;
    const int Qt = TT >> 4;
    if (gid >= BB * HH * Qt) return;                  // wave-uniform
    int qt = gid % Qt;
    int hh = (gid / Qt) % HH;
    int b  = gid / (Qt * HH);
    _Float16* lw = lds[w];

    FragH aq;                                         // Q fragment, loaded once
    const _Float16* qrow = q + ((size_t)(b * TT + qt * 16 + sub)) * CC + hh * HS;
    aq.h[0] = *(const v8h*)(qrow + half * 8);
    aq.h[1] = *(const v8h*)(qrow + 16 + half * 8);

    float m_run[8], l_run[8];
    v8f O0, O1;
    for (int r = 0; r < 8; r++) { m_run[r] = -1e30f; l_run[r] = 0.f; O0[r] = 0.f; O1[r] = 0.f; }
    const float sc = 0.08838834764831845f;            // (n_embd=128)^-0.5 per reference

    int nb = ((qt * 16 + 15) >> 5) + 1;               // causal: key blocks of 32
    for (int blk = 0; blk < nb; blk++) {
        int key0 = blk * 32;
        FragH bk0, bk1;
        const _Float16* kr0 = kk + ((size_t)(b * TT + key0 + sub)) * CC + hh * HS + half * 8;
        bk0.h[0] = *(const v8h*)kr0;  bk0.h[1] = *(const v8h*)(kr0 + 16);
        const _Float16* kr1 = kk + ((size_t)(b * TT + key0 + 16 + sub)) * CC + hh * HS + half * 8;
        bk1.h[0] = *(const v8h*)kr1;  bk1.h[1] = *(const v8h*)(kr1 + 16);

        v8f z; for (int r = 0; r < 8; r++) z[r] = 0.f;
        v8f S0 = wmma_f16(aq.v, bk0.v, z);            // scores, keys key0..+15
        v8f S1 = wmma_f16(aq.v, bk1.v, z);            // scores, keys key0+16..+31

        float p0[8], p1[8];
        for (int r = 0; r < 8; r++) {
            int qg = qt * 16 + half * 8 + r;          // C-layout: VGPR r -> row r+8*half
            p0[r] = (key0 + sub      <= qg) ? S0[r] * sc : -1e30f;
            p1[r] = (key0 + 16 + sub <= qg) ? S1[r] * sc : -1e30f;
        }
        for (int r = 0; r < 8; r++) {
            float rm = fmaxf(p0[r], p1[r]);           // row-max over 16 lanes of half
            for (int m = 8; m >= 1; m >>= 1) rm = fmaxf(rm, __shfl_xor(rm, m, 32));
            float mn = fmaxf(m_run[r], rm);
            float alpha = __expf(m_run[r] - mn);
            m_run[r] = mn;
            p0[r] = __expf(p0[r] - mn);
            p1[r] = __expf(p1[r] - mn);
            float rs = p0[r] + p1[r];
            for (int m = 8; m >= 1; m >>= 1) rs += __shfl_xor(rs, m, 32);
            l_run[r] = l_run[r] * alpha + rs;
            O0[r] *= alpha;  O1[r] *= alpha;
            lw[(half * 8 + r) * 32 + sub]      = (_Float16)p0[r];   // transpose via LDS
            lw[(half * 8 + r) * 32 + 16 + sub] = (_Float16)p1[r];
        }
        asm volatile("s_wait_dscnt 0" ::: "memory");  // cross-lane LDS visibility (same wave)

        FragH ap;                                     // P as A fragment (16x32, ld=32)
        ap.h[0] = *(const v8h*)(lw + sub * 32 + half * 8);
        ap.h[1] = *(const v8h*)(lw + sub * 32 + 16 + half * 8);
        FragH bv0, bv1;                               // V fragments from vt (contiguous t)
        const _Float16* vr0 = vt + ((size_t)(b * CC + hh * HS + sub)) * TT + key0 + half * 8;
        bv0.h[0] = *(const v8h*)vr0;  bv0.h[1] = *(const v8h*)(vr0 + 16);
        const _Float16* vr1 = vt + ((size_t)(b * CC + hh * HS + 16 + sub)) * TT + key0 + half * 8;
        bv1.h[0] = *(const v8h*)vr1;  bv1.h[1] = *(const v8h*)(vr1 + 16);
        O0 = wmma_f16(ap.v, bv0.v, O0);               // dims 0..15
        O1 = wmma_f16(ap.v, bv1.v, O1);               // dims 16..31
        asm volatile("s_wait_dscnt 0" ::: "memory");  // keep next-iter stores after reads
    }

    for (int r = 0; r < 8; r++) {
        float inv = 1.f / l_run[r];
        size_t orow = (size_t)(b * TT + qt * 16 + half * 8 + r) * CC + hh * HS;
        att[orow + sub]      = (_Float16)(O0[r] * inv);
        att[orow + 16 + sub] = (_Float16)(O1[r] * inv);
    }
}

// ---------------------------------------------------------------------------
// Host launch
// ---------------------------------------------------------------------------
static inline int cdiv(long a, long b) { return (int)((a + b - 1) / b); }

extern "C" void kernel_launch(void* const* d_in, const int* in_sizes, int n_in,
                              void* d_out, int out_size, void* d_ws, size_t ws_size,
                              hipStream_t stream) {
    const int*   idx     = (const int*)  d_in[0];
    const float* tok_emb = (const float*)d_in[1];
    const float* pos_emb = (const float*)d_in[2];
    const float* wq      = (const float*)d_in[3];
    const float* wk      = (const float*)d_in[4];
    const float* wv      = (const float*)d_in[5];
    const float* w_proj  = (const float*)d_in[6];
    const float* b_proj  = (const float*)d_in[7];
    const float* ln1_g   = (const float*)d_in[8];
    const float* ln1_b   = (const float*)d_in[9];
    const float* w_ff1   = (const float*)d_in[10];
    const float* b_ff1   = (const float*)d_in[11];
    const float* w_ff2   = (const float*)d_in[12];
    const float* b_ff2   = (const float*)d_in[13];
    const float* ln2_g   = (const float*)d_in[14];
    const float* ln2_b   = (const float*)d_in[15];
    const float* lnf_g   = (const float*)d_in[16];
    const float* lnf_b   = (const float*)d_in[17];
    const float* w_head  = (const float*)d_in[18];
    const float* b_head  = (const float*)d_in[19];
    float* logits = (float*)d_out;

    // ---- carve workspace (all sizes multiples of 256B) ----
    char* p = (char*)d_ws;
    auto carve = [&](size_t bytes) { char* r = p; p += bytes; return r; };
    float*     x     = (float*)    carve((size_t)BT * CC * 4);
    _Float16*  h16   = (_Float16*) carve((size_t)BT * CC * 2);
    _Float16*  q16   = (_Float16*) carve((size_t)BT * CC * 2);
    _Float16*  k16   = (_Float16*) carve((size_t)BT * CC * 2);
    _Float16*  vt16  = (_Float16*) carve((size_t)BT * CC * 2);
    _Float16*  att16 = (_Float16*) carve((size_t)BT * CC * 2);
    _Float16*  ff16  = (_Float16*) carve((size_t)BT * FF * 2);
    _Float16*  xf16  = (_Float16*) carve((size_t)BT * CC * 2);
    _Float16*  qT[LL], *kT[LL], *vT[LL], *pT[LL], *f1T[LL], *f2T[LL];
    for (int l = 0; l < LL; l++) {
        qT[l]  = (_Float16*)carve((size_t)CC * CC * 2);
        kT[l]  = (_Float16*)carve((size_t)CC * CC * 2);
        vT[l]  = (_Float16*)carve((size_t)CC * CC * 2);
        pT[l]  = (_Float16*)carve((size_t)CC * CC * 2);
        f1T[l] = (_Float16*)carve((size_t)FF * CC * 2);
        f2T[l] = (_Float16*)carve((size_t)CC * FF * 2);
    }
    _Float16* headT = (_Float16*)carve((size_t)VV * CC * 2);

    // ---- weight conversion (fp32 -> f16, transposed to [N,K]) ----
    for (int l = 0; l < LL; l++) {
        for (int h = 0; h < HH; h++) {
            size_t so = ((size_t)l * HH + h) * CC * HS;   // wq[l][h] : [C, hs]
            convt_kernel<<<cdiv(CC*HS,256),256,0,stream>>>(wq + so, qT[l] + (size_t)h*HS*CC, CC, HS);
            convt_kernel<<<cdiv(CC*HS,256),256,0,stream>>>(wk + so, kT[l] + (size_t)h*HS*CC, CC, HS);
            convt_kernel<<<cdiv(CC*HS,256),256,0,stream>>>(wv + so, vT[l] + (size_t)h*HS*CC, CC, HS);
        }
        convt_kernel<<<cdiv(CC*CC,256),256,0,stream>>>(w_proj + (size_t)l*CC*CC, pT[l],  CC, CC);
        convt_kernel<<<cdiv(CC*FF,256),256,0,stream>>>(w_ff1  + (size_t)l*CC*FF, f1T[l], CC, FF);
        convt_kernel<<<cdiv(FF*CC,256),256,0,stream>>>(w_ff2  + (size_t)l*FF*CC, f2T[l], FF, CC);
    }
    convt_kernel<<<cdiv(CC*VV,256),256,0,stream>>>(w_head, headT, CC, VV);

    // ---- embedding ----
    embed_kernel<<<cdiv((size_t)BT*32,256),256,0,stream>>>(idx, tok_emb, pos_emb, x);

    auto gemm_blocks = [&](int M, int N) {
        int Ng = ((N >> 4) + 3) >> 2;
        return cdiv((size_t)(M >> 4) * Ng, 8);
    };

    // ---- transformer layers ----
    for (int l = 0; l < LL; l++) {
        layernorm_kernel<<<cdiv(BT,8),256,0,stream>>>(x, ln1_g + l*CC, ln1_b + l*CC, h16, BT);
        gemm16_kernel<OUT_F16, false,false,false><<<gemm_blocks(BT,CC),256,0,stream>>>(
            h16, qT[l], nullptr, nullptr, q16, BT, CC, CC, 1);
        gemm16_kernel<OUT_F16, false,false,false><<<gemm_blocks(BT,CC),256,0,stream>>>(
            h16, kT[l], nullptr, nullptr, k16, BT, CC, CC, 1);
        gemm16_kernel<OUT_F16T,false,false,false><<<gemm_blocks(BT,CC),256,0,stream>>>(
            h16, vT[l], nullptr, nullptr, vt16, BT, CC, CC, TT);
        attn_kernel<<<cdiv(BB*HH*(TT/16),8),256,0,stream>>>(q16, k16, vt16, att16);
        gemm16_kernel<OUT_F32, false,true ,false><<<gemm_blocks(BT,CC),256,0,stream>>>(
            att16, pT[l], b_proj + l*CC, x, x, BT, CC, CC, 1);
        layernorm_kernel<<<cdiv(BT,8),256,0,stream>>>(x, ln2_g + l*CC, ln2_b + l*CC, h16, BT);
        gemm16_kernel<OUT_F16, true, false,false><<<gemm_blocks(BT,FF),256,0,stream>>>(
            h16, f1T[l], b_ff1 + l*FF, nullptr, ff16, BT, FF, CC, 1);
        gemm16_kernel<OUT_F32, false,true ,false><<<gemm_blocks(BT,CC),256,0,stream>>>(
            ff16, f2T[l], b_ff2 + l*CC, x, x, BT, CC, FF, 1);
    }

    // ---- final LN + LM head (NT stores: 655MB streaming output, never re-read) ----
    layernorm_kernel<<<cdiv(BT,8),256,0,stream>>>(x, lnf_g, lnf_b, xf16, BT);
    gemm16_kernel<OUT_F32, false,false,true><<<gemm_blocks(BT,VV),256,0,stream>>>(
        xf16, headT, b_head, nullptr, logits, BT, VV, CC, 1);
}
